// MultiHeadAttention_65360812310587
// MI455X (gfx1250) — matrix operand order repaired and checked by reading, not compiled, not run on hardware
//
#include <hip/hip_runtime.h>
#include <hip/hip_bf16.h>
#include <stdint.h>

// ---------------------------------------------------------------------------
// Types
// ---------------------------------------------------------------------------
typedef __attribute__((ext_vector_type(16))) __bf16 v16bf;
typedef __attribute__((ext_vector_type(8)))  __bf16 v8bf;
typedef __attribute__((ext_vector_type(4)))  __bf16 v4bf;
typedef __attribute__((ext_vector_type(8)))  float  v8f;
typedef __attribute__((ext_vector_type(4)))  unsigned int u32x4;
typedef __attribute__((ext_vector_type(8)))  int    i32x8;
typedef __attribute__((ext_vector_type(4)))  int    i32x4;

union FragU { v16bf v; v8bf h[2]; };

// ---------------------------------------------------------------------------
// WMMA fragment loaders (ISA 7.12.2 layouts), bf16 16x16x32
//   A (16x32): lane m=lane&15; half=lane>>4 -> K = {8h..8h+7, 16+8h..16+8h+7}
//   B (32x16): lane n=lane&15; half=lane>>4 -> K = {8h..8h+7, 16+8h..16+8h+7}
//   C (16x16 f32): reg r -> row r+8*half, col lane&15
// ---------------------------------------------------------------------------
__device__ inline v16bf frag_a_lds(const __bf16* base, int row0, int stride,
                                   int kofs, int lane) {
  const __bf16* p = base + (size_t)(row0 + (lane & 15)) * stride + kofs +
                    ((lane >> 4) << 3);
  FragU u;
  u.h[0] = *(const v8bf*)p;         // K = kbase .. kbase+7
  u.h[1] = *(const v8bf*)(p + 16);  // K = kbase+16 .. kbase+23
  return u.v;
}

__device__ inline v16bf frag_b_lds(const __bf16* base, int col0, int stride,
                                   int kofs, int lane) {
  const __bf16* p = base + (size_t)(col0 + (lane & 15)) * stride + kofs +
                    ((lane >> 4) << 3);
  FragU u;
  u.h[0] = *(const v8bf*)p;         // K = kbase .. kbase+7
  u.h[1] = *(const v8bf*)(p + 16);  // K = kbase+16 .. kbase+23
  return u.v;
}

// ---------------------------------------------------------------------------
// Half-wave (16-lane) reductions via ds_swizzle xor network (wave32).
// Masks < 16 keep lanes within their 16-lane half, matching the C-fragment
// row ownership (rows r in lanes 0-15, rows r+8 in lanes 16-31).
// ---------------------------------------------------------------------------
template <int M>
__device__ inline float swz_xor(float v) {
  return __int_as_float(
      __builtin_amdgcn_ds_swizzle(__float_as_int(v), 0x1F | (M << 10)));
}
__device__ inline float red_max16(float v) {
  v = fmaxf(v, swz_xor<1>(v));
  v = fmaxf(v, swz_xor<2>(v));
  v = fmaxf(v, swz_xor<4>(v));
  v = fmaxf(v, swz_xor<8>(v));
  return v;
}
__device__ inline float red_sum16(float v) {
  v += swz_xor<1>(v);
  v += swz_xor<2>(v);
  v += swz_xor<4>(v);
  v += swz_xor<8>(v);
  return v;
}

// Raw v_exp_f32: valid here because all inputs are <= 0 and finite
// (score - running_max, scaled); avoids libm's range-fixup sequence.
__device__ inline float fast_exp2(float x) {
  return __builtin_amdgcn_exp2f(x);
}

// ---------------------------------------------------------------------------
// fp32 -> bf16 convert (vectorized)
// ---------------------------------------------------------------------------
__global__ __launch_bounds__(256) void cvt_f32_bf16(
    const float* __restrict__ in, __bf16* __restrict__ out, long n) {
  long i = ((long)blockIdx.x * blockDim.x + threadIdx.x) * 4;
  if (i + 3 < n) {
    float4 f = *(const float4*)(in + i);
    v4bf o;
    o[0] = (__bf16)f.x; o[1] = (__bf16)f.y;
    o[2] = (__bf16)f.z; o[3] = (__bf16)f.w;
    *(v4bf*)(out + i) = o;
  }
}

// ---------------------------------------------------------------------------
// LDS-tiled transpose + convert: W (K x N, f32) -> Wt (N x K, bf16)
// ---------------------------------------------------------------------------
__global__ __launch_bounds__(256) void transpose_cvt_bf16(
    const float* __restrict__ W, __bf16* __restrict__ Wt, int K, int N) {
  __shared__ float t[32][33];
  const int n0 = blockIdx.x * 32, k0 = blockIdx.y * 32;
  const int lx = threadIdx.x & 31, ly = threadIdx.x >> 5;
#pragma unroll
  for (int i = 0; i < 32; i += 8)
    t[ly + i][lx] = W[(size_t)(k0 + ly + i) * N + n0 + lx];
  __syncthreads();
#pragma unroll
  for (int i = 0; i < 32; i += 8)
    Wt[(size_t)(n0 + ly + i) * K + k0 + lx] = (__bf16)t[lx][ly + i];
}

// ---------------------------------------------------------------------------
// bf16 WMMA GEMM: C[M,N] = A[M,K] * Bt[N,K]^T + bias
//   block = 256 thr (8 waves, 2x4), tile 128x128, K-step 32
//   TRANS_OUT: store C as [batch][col][s] (for V^T);  OUT_F32: fp32 output
// ---------------------------------------------------------------------------
template <bool TRANS_OUT, bool OUT_F32>
__global__ __launch_bounds__(256) void gemm_bf16_wmma(
    const __bf16* __restrict__ A, const __bf16* __restrict__ Bt,
    const float* __restrict__ bias, void* __restrict__ Cout,
    int M, int N, int K) {
  constexpr int LST = 48;  // LDS row stride (elts); 96B, multiple of 16B
  __shared__ __attribute__((aligned(16))) __bf16 As[128 * LST];
  __shared__ __attribute__((aligned(16))) __bf16 Bs[128 * LST];

  const int tid = threadIdx.x;
  const int lane = tid & 31, wave = tid >> 5;
  const int wm = wave >> 2, wn = wave & 3;  // 2 x 4 wave grid
  const int m0 = blockIdx.x * 128, n0 = blockIdx.y * 128;

  v8f acc[4][2] = {};

  for (int k0 = 0; k0 < K; k0 += 32) {
    // cooperative tile load: 128 rows x 32 cols each, 16B chunks
#pragma unroll
    for (int c = 0; c < 2; ++c) {
      const int idx = tid + c * 256;
      const int r = idx >> 2, ch = (idx & 3) << 3;
      *(v8bf*)(As + r * LST + ch) =
          *(const v8bf*)(A + (size_t)(m0 + r) * K + k0 + ch);
      *(v8bf*)(Bs + r * LST + ch) =
          *(const v8bf*)(Bt + (size_t)(n0 + r) * K + k0 + ch);
    }
    __syncthreads();

    v16bf b0 = frag_b_lds(Bs, wn * 32 + 0, LST, 0, lane);
    v16bf b1 = frag_b_lds(Bs, wn * 32 + 16, LST, 0, lane);
#pragma unroll
    for (int mi = 0; mi < 4; ++mi) {
      v16bf a = frag_a_lds(As, wm * 64 + mi * 16, LST, 0, lane);
      acc[mi][0] = __builtin_amdgcn_wmma_f32_16x16x32_bf16(
          false, a, false, b0, (short)0, acc[mi][0], false, false);
      acc[mi][1] = __builtin_amdgcn_wmma_f32_16x16x32_bf16(
          false, a, false, b1, (short)0, acc[mi][1], false, false);
    }
    __syncthreads();
  }

  const int hl = lane >> 4, c16 = lane & 15;
#pragma unroll
  for (int mi = 0; mi < 4; ++mi)
#pragma unroll
    for (int ni = 0; ni < 2; ++ni)
#pragma unroll
      for (int r = 0; r < 8; ++r) {
        const int row = m0 + wm * 64 + mi * 16 + r + hl * 8;
        const int col = n0 + wn * 32 + ni * 16 + c16;
        const float v = acc[mi][ni][r] + bias[col];
        if constexpr (OUT_F32) {
          ((float*)Cout)[(size_t)row * N + col] = v;
        } else if constexpr (TRANS_OUT) {
          const int b = row >> 10, s = row & 1023;  // S = 1024
          ((__bf16*)Cout)[((size_t)b * N + col) * 1024 + s] = (__bf16)v;
        } else {
          ((__bf16*)Cout)[(size_t)row * N + col] = (__bf16)v;
        }
      }
}

// ---------------------------------------------------------------------------
// Flash attention, one (batch, head) x 128-query block per workgroup.
//   Q[b,s,f] bf16, K[b,s,f] bf16, Vt[b,f,s] bf16, Feat[b,s,f] bf16
//   8 waves x 16 query rows; key blocks of 64; softmax(QK^T/sqrt(32)).
//   Q tile fetched by the Tensor Data Mover with pad (LDS row stride 144B).
// ---------------------------------------------------------------------------
__global__ __launch_bounds__(256) void attn_flash_wmma(
    const __bf16* __restrict__ Q, const __bf16* __restrict__ Kg,
    const __bf16* __restrict__ Vt, __bf16* __restrict__ Feat) {
  constexpr int LSA = 72;  // 144B row stride (128B data + 16B TDM pad)
  constexpr float CEXP = 1.4426950408889634f / 5.656854249492381f;  // log2e/sqrt(32)
  __shared__ __attribute__((aligned(16))) __bf16 Qs[128 * LSA];
  __shared__ __attribute__((aligned(16))) __bf16 Ks[64 * LSA];
  __shared__ __attribute__((aligned(16))) __bf16 Vs[64 * LSA];
  __shared__ __attribute__((aligned(16))) __bf16 Ps[128 * LSA];

  const int tid = threadIdx.x;
  const int lane = tid & 31, wave = tid >> 5;
  const int hl = lane >> 4, c16 = lane & 15;
  const int qb = blockIdx.x * 128;
  const int head = blockIdx.y & 31;
  const int bb = blockIdx.y >> 5;

  // --- TDM: async copy the 128x64 Q tile (global row stride 2048 elts)
  //     into LDS; pad_interval = 32 dwords (one 128B row), pad = 4 dwords
  //     -> LDS row stride 144B = LSA elements. Issued once by wave 0.
  if (tid < 32) {
    const unsigned long long ga = (unsigned long long)(uintptr_t)(
        Q + ((size_t)(bb * 1024 + qb)) * 2048 + head * 64);
    const unsigned int lds = (unsigned int)(uintptr_t)(&Qs[0]);
    u32x4 g0 = {0u, 0u, 0u, 0u};
    g0[0] = 1u;                    // count=1, user descriptor
    g0[1] = lds;                   // lds_addr
    g0[2] = (unsigned int)ga;      // global_addr[31:0]
    g0[3] = (unsigned int)((ga >> 32) & 0x1FFFFFFu) | (2u << 30);  // type=2
    i32x8 g1 = {0, 0, 0, 0, 0, 0, 0, 0};
    g1[0] = (1 << 16) | (1 << 20) | (4 << 22) | (3 << 25);
    //        data=2B   pad_en      intvl=32dw   amount=4dw
    g1[1] = (2048 & 0xFFFF) << 16;               // tensor_dim0 lo16
    g1[2] = (2048 >> 16) | ((4096 & 0xFFFF) << 16);  // dim0 hi | dim1 lo
    g1[3] = (4096 >> 16) | (64 << 16);           // dim1 hi | tile_dim0=64
    g1[4] = 128;                                 // tile_dim1=128, tile_dim2=0
    g1[5] = 2048;                                // tensor_dim0_stride lo32
    g1[6] = ((2048 & 0xFFFF) << 16);             // stride0 hi | stride1 lo
    g1[7] = (2048 >> 16);                        // stride1 hi
    i32x4 gz4 = {0, 0, 0, 0};
    i32x8 gz8 = {0, 0, 0, 0, 0, 0, 0, 0};
    __builtin_amdgcn_tensor_load_to_lds(g0, g1, gz4, gz4, gz8, 0);
    __builtin_amdgcn_s_wait_tensorcnt(0);
  }

  float m_i[8], l_i[8];
#pragma unroll
  for (int r = 0; r < 8; ++r) { m_i[r] = -3.0e38f; l_i[r] = 0.f; }
  v8f acc[4] = {};

  for (int kb = 0; kb < 1024; kb += 64) {
    // cooperative K / V^T tile loads (64 x 64 each), 16B chunks
#pragma unroll
    for (int c = 0; c < 2; ++c) {
      const int idx = tid + c * 256;
      const int r = idx >> 3, ch = (idx & 7) << 3;
      *(v8bf*)(Ks + r * LSA + ch) = *(const v8bf*)(
          Kg + ((size_t)(bb * 1024 + kb + r)) * 2048 + head * 64 + ch);
      *(v8bf*)(Vs + r * LSA + ch) = *(const v8bf*)(
          Vt + ((size_t)(bb * 2048 + head * 64 + r)) * 1024 + kb + ch);
    }
    if (kb + 64 < 1024) {  // hint next tiles toward the caches
      __builtin_prefetch(
          Kg + ((size_t)(bb * 1024 + kb + 64 + (tid >> 3))) * 2048 + head * 64,
          0, 1);
      __builtin_prefetch(
          Vt + ((size_t)(bb * 2048 + head * 64 + (tid >> 3))) * 1024 + kb + 64,
          0, 1);
    }
    __syncthreads();

    // --- scores tile: S = Q * K^T  (16 rows x 64 keys per wave)
    v8f s[4] = {};
#pragma unroll
    for (int ks = 0; ks < 64; ks += 32) {
      v16bf aq = frag_a_lds(Qs, wave * 16, LSA, ks, lane);
#pragma unroll
      for (int nt = 0; nt < 4; ++nt) {
        v16bf bkf = frag_b_lds(Ks, nt * 16, LSA, ks, lane);
        s[nt] = __builtin_amdgcn_wmma_f32_16x16x32_bf16(
            false, aq, false, bkf, (short)0, s[nt], false, false);
      }
    }

    // --- online softmax (scale folded into exp2 constant)
#pragma unroll
    for (int r = 0; r < 8; ++r) {
      float mx = fmaxf(fmaxf(s[0][r], s[1][r]), fmaxf(s[2][r], s[3][r]));
      mx = red_max16(mx);
      const float mn = fmaxf(m_i[r], mx);
      const float alpha = fast_exp2((m_i[r] - mn) * CEXP);
      float rs = 0.f;
#pragma unroll
      for (int nt = 0; nt < 4; ++nt) {
        const float p = fast_exp2((s[nt][r] - mn) * CEXP);
        s[nt][r] = p;
        rs += p;
      }
      rs = red_sum16(rs);
      l_i[r] = l_i[r] * alpha + rs;
      m_i[r] = mn;
#pragma unroll
      for (int nt = 0; nt < 4; ++nt) acc[nt][r] *= alpha;
    }

    // --- stage P (bf16) in LDS so it can be re-read in A-fragment layout
#pragma unroll
    for (int nt = 0; nt < 4; ++nt)
#pragma unroll
      for (int r = 0; r < 8; ++r)
        Ps[(size_t)(wave * 16 + r + hl * 8) * LSA + nt * 16 + c16] =
            (__bf16)s[nt][r];

    // --- O += P * V  (K-dim = 64 keys, N = 64 head channels)
#pragma unroll
    for (int ks = 0; ks < 64; ks += 32) {
      v16bf ap = frag_a_lds(Ps, wave * 16, LSA, ks, lane);
#pragma unroll
      for (int nt = 0; nt < 4; ++nt) {
        v16bf bv = frag_b_lds(Vs, nt * 16, LSA, ks, lane);
        acc[nt] = __builtin_amdgcn_wmma_f32_16x16x32_bf16(
            false, ap, false, bv, (short)0, acc[nt], false, false);
      }
    }
    __syncthreads();
  }

  // --- normalize and store feats row-major [B*S, 2048]
#pragma unroll
  for (int r = 0; r < 8; ++r) {
    const float inv = __builtin_amdgcn_rcpf(l_i[r]);
    const int row = bb * 1024 + qb + wave * 16 + r + hl * 8;
#pragma unroll
    for (int nt = 0; nt < 4; ++nt)
      Feat[(size_t)row * 2048 + head * 64 + nt * 16 + c16] =
          (__bf16)(acc[nt][r] * inv);
  }
}

// ---------------------------------------------------------------------------
// Launch: cvt -> 4x transpose -> QKV GEMMs -> flash attention -> out GEMM
// ---------------------------------------------------------------------------
extern "C" void kernel_launch(void* const* d_in, const int* in_sizes, int n_in,
                              void* d_out, int out_size, void* d_ws,
                              size_t ws_size, hipStream_t stream) {
  (void)in_sizes; (void)n_in; (void)out_size; (void)ws_size;
  const float* x  = (const float*)d_in[0];
  const float* Wq = (const float*)d_in[1];
  const float* bq = (const float*)d_in[2];
  const float* Wk = (const float*)d_in[3];
  const float* bk = (const float*)d_in[4];
  const float* Wv = (const float*)d_in[5];
  const float* bv = (const float*)d_in[6];
  const float* Wo = (const float*)d_in[7];
  const float* bo = (const float*)d_in[8];
  float* out = (float*)d_out;

  char* ws = (char*)d_ws;
  const size_t SZ_X = (size_t)4096 * 2048 * 2;  // bf16 activations
  const size_t SZ_W = (size_t)2048 * 2048 * 2;  // bf16 weight
  __bf16* xb  = (__bf16*)(ws);
  __bf16* wqT = (__bf16*)(ws + SZ_X);
  __bf16* wkT = (__bf16*)(ws + SZ_X + 1 * SZ_W);
  __bf16* wvT = (__bf16*)(ws + SZ_X + 2 * SZ_W);
  __bf16* woT = (__bf16*)(ws + SZ_X + 3 * SZ_W);
  __bf16* Qb  = (__bf16*)(ws + 1 * SZ_X + 4 * SZ_W);
  __bf16* Kb  = (__bf16*)(ws + 2 * SZ_X + 4 * SZ_W);
  __bf16* Vtb = (__bf16*)(ws + 3 * SZ_X + 4 * SZ_W);
  __bf16* Fb  = xb;  // reuse x region: x is dead after the QKV GEMMs

  const long NX = (long)4096 * 2048;
  cvt_f32_bf16<<<(unsigned)((NX / 4 + 255) / 256), 256, 0, stream>>>(x, xb, NX);

  dim3 tg(2048 / 32, 2048 / 32);
  transpose_cvt_bf16<<<tg, 256, 0, stream>>>(Wq, wqT, 2048, 2048);
  transpose_cvt_bf16<<<tg, 256, 0, stream>>>(Wk, wkT, 2048, 2048);
  transpose_cvt_bf16<<<tg, 256, 0, stream>>>(Wv, wvT, 2048, 2048);
  transpose_cvt_bf16<<<tg, 256, 0, stream>>>(Wo, woT, 2048, 2048);

  dim3 gg(4096 / 128, 2048 / 128);
  gemm_bf16_wmma<false, false><<<gg, 256, 0, stream>>>(xb, wqT, bq, Qb, 4096, 2048, 2048);
  gemm_bf16_wmma<false, false><<<gg, 256, 0, stream>>>(xb, wkT, bk, Kb, 4096, 2048, 2048);
  gemm_bf16_wmma<true,  false><<<gg, 256, 0, stream>>>(xb, wvT, bv, Vtb, 4096, 2048, 2048);

  attn_flash_wmma<<<dim3(1024 / 128, 4 * 32), 256, 0, stream>>>(Qb, Kb, Vtb, Fb);

  gemm_bf16_wmma<false, true><<<gg, 256, 0, stream>>>(Fb, woT, bo, out, 4096, 2048, 2048);
}
